// ProjectionV2_74208444940728
// MI455X (gfx1250) — matrix operand order repaired and checked by reading, not compiled, not run on hardware
//
#include <hip/hip_runtime.h>
#include <hip/hip_bf16.h>

typedef __attribute__((ext_vector_type(16))) _Float16 v16h;
typedef __attribute__((ext_vector_type(8)))  _Float16 v8h;
typedef __attribute__((ext_vector_type(8)))  float    v8f;

#define DEV __device__ __forceinline__

DEV float gelu_tanh(float x){
    float x3 = x*x*x;
    return 0.5f*x*(1.0f + tanhf(0.7978845608028654f*(x + 0.044715f*x3)));
}

// ---------------- elementwise / prep ----------------
__global__ void k_f2h(const float* __restrict__ in, _Float16* __restrict__ out, int n){
    int i = blockIdx.x*256 + threadIdx.x;
    if (i < n) out[i] = (_Float16)in[i];
}
__global__ void k_fcwpad(const float* __restrict__ fcw, _Float16* __restrict__ out){
    int i = blockIdx.x*256 + threadIdx.x;           // 800*768 entries
    if (i >= 800*768) return;
    int r = i/768, c = i - r*768;
    out[i] = (r < 784) ? (_Float16)fcw[r*768 + c] : (_Float16)0.0f;
}
__global__ void k_rep_proto(const float* __restrict__ proto, float* __restrict__ out){
    int i = blockIdx.x*256 + threadIdx.x;           // 96*768 entries
    if (i >= 96*768) return;
    int r = i/768, c = i - r*768;
    out[i] = proto[(r%6)*768 + c];
}
__global__ void k_zero(float* a){ if (threadIdx.x < 8) a[threadIdx.x] = 0.0f; }

// ---------------- bilinear 4x downsample: x[B,64,64,C] -> x0[B,16,16,C] ----------------
// align_corners=False, scale 4: sample coord = 4i+1.5 -> mean of pixels 4i+1,4i+2
__global__ __launch_bounds__(256) void k_down(const float* __restrict__ x, float* __restrict__ x0){
    int row = blockIdx.x;                 // b*256 + i*16 + j
    int b = row >> 8, p = row & 255;
    int i = p >> 4, j = p & 15;
    const float* base = x + (size_t)b*4096*768;
    int t00 = (4*i+1)*64 + (4*j+1);
    const float* p00 = base + (size_t)t00*768;
    const float* p01 = p00 + 768;
    const float* p10 = p00 + 64*768;
    const float* p11 = p10 + 768;
    float* o = x0 + (size_t)row*768;
    for (int c = threadIdx.x; c < 768; c += 256)
        o[c] = 0.25f*(p00[c] + p01[c] + p10[c] + p11[c]);
}

// ---------------- LayerNorm over D=768, f32 in -> f16 out ----------------
__global__ __launch_bounds__(256) void k_ln(const float* __restrict__ in, const float* __restrict__ g,
                                            const float* __restrict__ bt, _Float16* __restrict__ out){
    int row = blockIdx.x;
    const float* p = in + (size_t)row*768;
    float s = 0.f, s2 = 0.f;
    for (int c = threadIdx.x; c < 768; c += 256){ float v = p[c]; s += v; s2 += v*v; }
    __shared__ float rs[256], rq[256];
    int t = threadIdx.x;
    rs[t] = s; rq[t] = s2; __syncthreads();
    for (int st = 128; st >= 1; st >>= 1){
        if (t < st){ rs[t] += rs[t+st]; rq[t] += rq[t+st]; }
        __syncthreads();
    }
    float mean = rs[0]*(1.0f/768.0f);
    float var  = rq[0]*(1.0f/768.0f) - mean*mean;
    float rstd = rsqrtf(var + 1e-8f);
    _Float16* o = out + (size_t)row*768;
    for (int c = t; c < 768; c += 256) o[c] = (_Float16)((p[c]-mean)*rstd*g[c] + bt[c]);
}

// ---------------- WMMA GEMM: C[M,N] = A[M,K](f16) @ B[K,N](f16) + epilogue ----------------
constexpr int EB = 1;  // add bias[N]
constexpr int EG = 2;  // tanh-gelu
constexpr int ER = 4;  // add f32 residual [M,N]
constexpr int EH = 8;  // store f16 (else f32)

// FULLM: M is a multiple of 128 -> no row guards, and A tile staged with
// CDNA5 async direct-to-LDS loads (ASYNCcnt) instead of VGPR round-trips.
template<int EPI, bool FULLM>
__global__ __launch_bounds__(256)
void k_gemm(const _Float16* __restrict__ A, const _Float16* __restrict__ Bw,
            const float* __restrict__ bias, const float* __restrict__ res,
            void* __restrict__ outp, int M, int N, int K){
    constexpr int BM = 128, BN = 128, BK = 32, LDK = BK + 8;
    __shared__ __align__(16) _Float16 As[BM][LDK];
    __shared__ __align__(16) _Float16 Bs[BN][LDK];     // B stored transposed: Bs[n][k]
    int tid  = threadIdx.x;
    int bm   = blockIdx.y, bn = blockIdx.x;
    int wid  = tid >> 5, lane = tid & 31;
    int wm   = wid & 1, wn = wid >> 1;                 // 2x4 wave grid, wave tile 64x32
    int half = (lane >> 4) & 1, l16 = lane & 15;

    v8f acc[4][2];
    #pragma unroll
    for (int mi = 0; mi < 4; mi++)
        #pragma unroll
        for (int ni = 0; ni < 2; ni++){ v8f z = {}; acc[mi][ni] = z; }

    for (int k0 = 0; k0 < K; k0 += BK){
        // ---- stage A tile (row major): 512 slots of 8 f16, 2 per thread ----
        if (FULLM){
            // async global->LDS, 16B per lane per op; tracked by ASYNCcnt
            #pragma unroll
            for (int it = 0; it < 2; it++){
                int s = tid + it*256;
                int r = s >> 2, seg = s & 3;
                const _Float16* gp = A + (size_t)(bm*BM + r)*K + k0 + seg*8;
                unsigned lds = (unsigned)(unsigned long long)&As[r][seg*8];
                asm volatile("global_load_async_to_lds_b128 %0, %1, off"
                             :: "v"(lds), "v"((unsigned long long)gp) : "memory");
            }
        } else {
            #pragma unroll
            for (int it = 0; it < 2; it++){
                int s = tid + it*256;
                int r = s >> 2, seg = s & 3;
                int row = bm*BM + r;
                uint4 val = make_uint4(0u,0u,0u,0u);
                if (row < M) val = *(const uint4*)(A + (size_t)row*K + k0 + seg*8);
                *(uint4*)&As[r][seg*8] = val;
            }
        }
        // ---- stage B tile transposed ----
        #pragma unroll
        for (int it = 0; it < 2; it++){
            int s = tid + it*256;
            int kr = s >> 4, nseg = s & 15;
            const _Float16* bp = Bw + (size_t)(k0+kr)*N + bn*BN + nseg*8;
            v8h bv = *(const v8h*)bp;
            #pragma unroll
            for (int i2 = 0; i2 < 8; i2++) Bs[nseg*8 + i2][kr] = bv[i2];
        }
        if (k0 + BK < K){
            __builtin_prefetch(A  + (size_t)(bm*BM + (tid >> 1))*K + k0 + BK, 0, 1);
            __builtin_prefetch(Bw + (size_t)(k0 + BK + (tid & 31))*N + bn*BN, 0, 1);
        }
        if (FULLM)
            asm volatile("s_wait_asynccnt 0x0" ::: "memory");
        __syncthreads();

        v16h af[4], bf[2];
        #pragma unroll
        for (int mi = 0; mi < 4; mi++){
            int r = wm*64 + mi*16 + l16;
            v8h lo = *(const v8h*)&As[r][half*8];       // K = half*8 .. +7
            v8h hi = *(const v8h*)&As[r][half*8 + 16];  // K = half*8+16 .. +23
            af[mi] = __builtin_shufflevector(lo, hi, 0,1,2,3,4,5,6,7,8,9,10,11,12,13,14,15);
        }
        #pragma unroll
        for (int ni = 0; ni < 2; ni++){
            int nn = wn*32 + ni*16 + l16;
            v8h lo = *(const v8h*)&Bs[nn][half*8];      // K = half*8 .. +7
            v8h hi = *(const v8h*)&Bs[nn][half*8 + 16]; // K = half*8+16 .. +23
            bf[ni] = __builtin_shufflevector(lo, hi, 0,1,2,3,4,5,6,7,8,9,10,11,12,13,14,15);
        }
        #pragma unroll
        for (int mi = 0; mi < 4; mi++)
            #pragma unroll
            for (int ni = 0; ni < 2; ni++)
                acc[mi][ni] = __builtin_amdgcn_wmma_f32_16x16x32_f16(
                    false, af[mi], false, bf[ni], (short)0, acc[mi][ni], false, false);
        __syncthreads();
    }

    // epilogue: lane holds col = l16, rows r2 + 8*half
    #pragma unroll
    for (int mi = 0; mi < 4; mi++){
        #pragma unroll
        for (int ni = 0; ni < 2; ni++){
            int col = bn*BN + wn*32 + ni*16 + l16;
            #pragma unroll
            for (int r2 = 0; r2 < 8; r2++){
                int row = bm*BM + wm*64 + mi*16 + half*8 + r2;
                if (FULLM || row < M){
                    float v = acc[mi][ni][r2];
                    if (EPI & EB) v += bias[col];
                    if (EPI & EG) v = gelu_tanh(v);
                    if (EPI & ER) v += res[(size_t)row*N + col];
                    if (EPI & EH) ((_Float16*)outp)[(size_t)row*N + col] = (_Float16)v;
                    else          ((float*)outp)[(size_t)row*N + col] = v;
                }
            }
        }
    }
}

// ---------------- aggregation attention: 6 queries x 4096 keys, per (b,h) ----------------
__global__ __launch_bounds__(384)
void k_agg_attn(const _Float16* __restrict__ qh, const _Float16* __restrict__ kv,
                _Float16* __restrict__ o){
    int b = blockIdx.x / 12, h = blockIdx.x % 12;
    __shared__ _Float16 qs[6][64];
    __shared__ float sc[6][512];
    __shared__ float red[384];
    __shared__ float mq[6], dq[6], scl[6];
    int tid = threadIdx.x;
    int q = tid >> 6, d = tid & 63;                   // 384 = 6*64 exactly
    qs[q][d] = qh[q*768 + h*64 + d];
    if (tid < 6){ mq[tid] = -1e30f; dq[tid] = 0.f; scl[tid] = 0.f; }
    float acc = 0.f;
    __syncthreads();
    for (int t0 = 0; t0 < 4096; t0 += 512){
        for (int t = tid; t < 512; t += 384){
            const _Float16* kp = kv + (size_t)(b*4096 + t0 + t)*1536 + h*64;
            float s[6] = {0,0,0,0,0,0};
            for (int c = 0; c < 64; c++){
                float kc = (float)kp[c];
                #pragma unroll
                for (int qi = 0; qi < 6; qi++) s[qi] += kc * (float)qs[qi][c];
            }
            #pragma unroll
            for (int qi = 0; qi < 6; qi++) sc[qi][t] = s[qi]*0.125f;
        }
        __syncthreads();
        // chunk max per q (64-thread groups)
        float m = -1e30f;
        for (int j = d; j < 512; j += 64) m = fmaxf(m, sc[q][j]);
        red[tid] = m; __syncthreads();
        for (int st = 32; st >= 1; st >>= 1){ if (d < st) red[tid] = fmaxf(red[tid], red[tid+st]); __syncthreads(); }
        if (d == 0){ float nm = fmaxf(mq[q], red[q*64]); scl[q] = __expf(mq[q] - nm); mq[q] = nm; }
        __syncthreads();
        float ssum = 0.f;
        for (int j = d; j < 512; j += 64){ float pv = __expf(sc[q][j] - mq[q]); sc[q][j] = pv; ssum += pv; }
        red[tid] = ssum; __syncthreads();
        for (int st = 32; st >= 1; st >>= 1){ if (d < st) red[tid] += red[tid+st]; __syncthreads(); }
        if (d == 0) dq[q] = dq[q]*scl[q] + red[q*64];
        __syncthreads();
        acc *= scl[q];
        const _Float16* vbase = kv + (size_t)(b*4096 + t0)*1536 + 768 + h*64;
        for (int t = 0; t < 512; t++)
            acc += sc[q][t] * (float)vbase[(size_t)t*1536 + d];
        __syncthreads();
    }
    o[(size_t)(b*6 + q)*768 + h*64 + d] = (_Float16)(acc / dq[q]);
}

// ---------------- decoder attention: 4096 tokens x 6 keys, thread per (token,head) ----------------
__global__ __launch_bounds__(256)
void k_dec_attn(const _Float16* __restrict__ qh, const _Float16* __restrict__ kv,
                _Float16* __restrict__ o){
    int idx = blockIdx.x*256 + threadIdx.x;
    if (idx >= 4096*12) return;
    int token = idx / 12, h = idx % 12;
    int b = token >> 8;
    const _Float16* qp = qh + (size_t)token*768 + h*64;
    float qr[64];
    #pragma unroll
    for (int c = 0; c < 64; c++) qr[c] = (float)qp[c];
    float s[6]; float m = -1e30f;
    #pragma unroll
    for (int j = 0; j < 6; j++){
        const _Float16* kp = kv + (size_t)(b*6 + j)*1536 + h*64;
        float a = 0.f;
        for (int c = 0; c < 64; c++) a += qr[c]*(float)kp[c];
        s[j] = a*0.125f; m = fmaxf(m, s[j]);
    }
    float den = 0.f;
    #pragma unroll
    for (int j = 0; j < 6; j++){ s[j] = __expf(s[j]-m); den += s[j]; }
    float inv = 1.0f/den;
    float orr[64];
    #pragma unroll
    for (int c = 0; c < 64; c++) orr[c] = 0.f;
    #pragma unroll
    for (int j = 0; j < 6; j++){
        const _Float16* vp = kv + (size_t)(b*6 + j)*1536 + 768 + h*64;
        float w = s[j]*inv;
        for (int c = 0; c < 64; c++) orr[c] += w*(float)vp[c];
    }
    _Float16* op = o + (size_t)token*768 + h*64;
    for (int c = 0; c < 64; c++) op[c] = (_Float16)orr[c];
}

// ---------------- gather loss: per x0 token, min over 6 prototypes of (1 - cos) ----------------
__global__ __launch_bounds__(256)
void k_gather(const float* __restrict__ x0, const float* __restrict__ agg, float* __restrict__ gacc){
    int row = blockIdx.x;               // b*256 + t
    int b = row >> 8;
    __shared__ float xr[768];
    __shared__ float r0[256], r1[256];
    int t = threadIdx.x;
    for (int c = t; c < 768; c += 256) xr[c] = x0[(size_t)row*768 + c];
    __syncthreads();
    float nx = 0.f;
    for (int c = t; c < 768; c += 256) nx += xr[c]*xr[c];
    r0[t] = nx; __syncthreads();
    for (int st = 128; st >= 1; st >>= 1){ if (t < st) r0[t] += r0[t+st]; __syncthreads(); }
    float nxv = fmaxf(sqrtf(r0[0]), 1e-8f);
    __syncthreads();
    float best = 1e30f;
    for (int i = 0; i < 6; i++){
        const float* ap = agg + (size_t)(b*6 + i)*768;
        float dt = 0.f, na = 0.f;
        for (int c = t; c < 768; c += 256){ float av = ap[c]; dt += av*xr[c]; na += av*av; }
        r0[t] = dt; r1[t] = na; __syncthreads();
        for (int st = 128; st >= 1; st >>= 1){ if (t < st){ r0[t]+=r0[t+st]; r1[t]+=r1[t+st]; } __syncthreads(); }
        float cosv = r0[0] / (nxv * fmaxf(sqrtf(r1[0]), 1e-8f));
        best = fminf(best, 1.0f - cosv);
        __syncthreads();
    }
    if (t == 0) atomicAdd(gacc, best);
}

// ---------------- per-batch global cosine (sm loss) ----------------
__global__ __launch_bounds__(256)
void k_sm(const float* __restrict__ en, const float* __restrict__ de, float* __restrict__ smacc){
    int b = blockIdx.x;
    const float* a = en + (size_t)b*256*768;
    const float* d = de + (size_t)b*256*768;
    __shared__ float r0[256], r1[256], r2[256];
    int t = threadIdx.x;
    float dt = 0.f, na = 0.f, nd = 0.f;
    for (int n = t; n < 256*768; n += 256){ float av = a[n], dv = d[n]; dt += av*dv; na += av*av; nd += dv*dv; }
    r0[t]=dt; r1[t]=na; r2[t]=nd; __syncthreads();
    for (int st = 128; st >= 1; st >>= 1){
        if (t < st){ r0[t]+=r0[t+st]; r1[t]+=r1[t+st]; r2[t]+=r2[t+st]; }
        __syncthreads();
    }
    if (t == 0) atomicAdd(smacc, 1.0f - r0[0]/fmaxf(sqrtf(r1[0])*sqrtf(r2[0]), 1e-8f));
}

// ---------------- channel cosine map per token ----------------
__global__ __launch_bounds__(256)
void k_cosmap(const float* __restrict__ en, const float* __restrict__ de, float* __restrict__ m){
    int row = blockIdx.x;               // 4096
    const float* a = en + (size_t)row*768;
    const float* bb = de + (size_t)row*768;
    __shared__ float r0[256], r1[256], r2[256];
    int t = threadIdx.x;
    float dt = 0.f, na = 0.f, nb = 0.f;
    for (int c = t; c < 768; c += 256){ float av = a[c], bv = bb[c]; dt += av*bv; na += av*av; nb += bv*bv; }
    r0[t]=dt; r1[t]=na; r2[t]=nb; __syncthreads();
    for (int st = 128; st >= 1; st >>= 1){
        if (t < st){ r0[t]+=r0[t+st]; r1[t]+=r1[t+st]; r2[t]+=r2[t+st]; }
        __syncthreads();
    }
    if (t == 0) m[row] = r0[0] / fmaxf(sqrtf(r1[0])*sqrtf(r2[0]), 1e-8f);
}

// ---------------- bilinear 16->64 upsample of m + build concat matrix [65536, 800] f16 ----------------
__global__ __launch_bounds__(256)
void k_upcat(const float* __restrict__ x, const float* __restrict__ m, _Float16* __restrict__ cat){
    int row = blockIdx.x;               // b*4096 + I*64 + J
    int b = row >> 12, p = row & 4095;
    int I = p >> 6, J = p & 63;
    float sy = (I + 0.5f)*0.25f - 0.5f;
    float sx = (J + 0.5f)*0.25f - 0.5f;
    int y0 = (int)floorf(sy), x0i = (int)floorf(sx);
    float fy = sy - (float)y0, fx = sx - (float)x0i;
    int y0c = y0 < 0 ? 0 : (y0 > 15 ? 15 : y0);
    int y1c = (y0+1) < 0 ? 0 : ((y0+1) > 15 ? 15 : y0+1);
    int x0c = x0i < 0 ? 0 : (x0i > 15 ? 15 : x0i);
    int x1c = (x0i+1) < 0 ? 0 : ((x0i+1) > 15 ? 15 : x0i+1);
    const float* mb = m + b*256;
    float mv = (1.f-fy)*((1.f-fx)*mb[y0c*16 + x0c] + fx*mb[y0c*16 + x1c])
             +      fy *((1.f-fx)*mb[y1c*16 + x0c] + fx*mb[y1c*16 + x1c]);
    const float* xp = x + (size_t)row*768;
    _Float16* cp = cat + (size_t)row*800;
    for (int c = threadIdx.x; c < 800; c += 256){
        float v;
        if (c < 768)      v = xp[c];
        else if (c < 784) v = mv;
        else              v = 0.f;
        cp[c] = (_Float16)v;
    }
}

__global__ void k_final(const float* __restrict__ accs, float* __restrict__ out){
    if (threadIdx.x == 0 && blockIdx.x == 0){
        out[(size_t)65536*768]     = 0.2f * accs[0] * (1.0f/4096.0f);  // mean over B*256, x0.2
        out[(size_t)65536*768 + 1] = accs[1] * (1.0f/16.0f);           // mean over B
    }
}

extern "C" void kernel_launch(void* const* d_in, const int* in_sizes, int n_in,
                              void* d_out, int out_size, void* d_ws, size_t ws_size,
                              hipStream_t stream){
    (void)in_sizes; (void)n_in; (void)out_size; (void)ws_size;
    const float* x     = (const float*)d_in[0];
    const float* proto = (const float*)d_in[1];
    const float* fcw   = (const float*)d_in[4];
    const float* fcb   = (const float*)d_in[5];
    const float *ag1=(const float*)d_in[6],  *ab1=(const float*)d_in[7];
    const float *awq=(const float*)d_in[8],  *abq=(const float*)d_in[9];
    const float *awkv=(const float*)d_in[10],*abkv=(const float*)d_in[11];
    const float *awo=(const float*)d_in[12], *abo=(const float*)d_in[13];
    const float *ag2=(const float*)d_in[14], *ab2=(const float*)d_in[15];
    const float *aw1=(const float*)d_in[16], *ac1=(const float*)d_in[17];
    const float *aw2=(const float*)d_in[18], *ac2=(const float*)d_in[19];
    const float *dg1=(const float*)d_in[20], *db1=(const float*)d_in[21];
    const float *dwq=(const float*)d_in[22], *dbq=(const float*)d_in[23];
    const float *dwkv=(const float*)d_in[24],*dbkv=(const float*)d_in[25];
    const float *dwo=(const float*)d_in[26], *dbo=(const float*)d_in[27];
    const float *dg2=(const float*)d_in[28], *db2=(const float*)d_in[29];
    const float *dw1=(const float*)d_in[30], *dc1=(const float*)d_in[31];
    const float *dw2=(const float*)d_in[32], *dc2=(const float*)d_in[33];
    float* out = (float*)d_out;

    char* W = (char*)d_ws;
    size_t off = 0;
    auto alloc = [&](size_t bytes)->char*{
        char* p = W + off; off += (bytes + 255) & ~(size_t)255; return p;
    };
    _Float16* xn    = (_Float16*)alloc((size_t)65536*768*2);
    _Float16* kvA   = (_Float16*)alloc((size_t)65536*1536*2);
    float*    x0    = (float*)   alloc((size_t)4096*768*4);
    _Float16* pln   = (_Float16*)alloc((size_t)6*768*2);
    _Float16* qhA   = (_Float16*)alloc((size_t)6*768*2);
    _Float16* oA    = (_Float16*)alloc((size_t)96*768*2);
    float*    prep  = (float*)   alloc((size_t)96*768*4);
    float*    hA    = (float*)   alloc((size_t)96*768*4);
    _Float16* hnA   = (_Float16*)alloc((size_t)96*768*2);
    _Float16* f1A   = (_Float16*)alloc((size_t)96*3072*2);
    float*    agg   = (float*)   alloc((size_t)96*768*4);
    _Float16* aggn  = (_Float16*)alloc((size_t)96*768*2);
    _Float16* x0n   = (_Float16*)alloc((size_t)4096*768*2);
    _Float16* qhD   = (_Float16*)alloc((size_t)4096*768*2);
    _Float16* kvD   = (_Float16*)alloc((size_t)96*1536*2);
    _Float16* oD    = (_Float16*)alloc((size_t)4096*768*2);
    float*    hD    = (float*)   alloc((size_t)4096*768*4);
    _Float16* hnD   = (_Float16*)alloc((size_t)4096*768*2);
    _Float16* f1D   = (_Float16*)alloc((size_t)4096*3072*2);
    float*    de    = (float*)   alloc((size_t)4096*768*4);
    float*    mtok  = (float*)   alloc((size_t)4096*4);
    float*    accs  = (float*)   alloc(256);
    _Float16* wq16a = (_Float16*)alloc((size_t)768*768*2);
    _Float16* wkv16a= (_Float16*)alloc((size_t)768*1536*2);
    _Float16* wo16a = (_Float16*)alloc((size_t)768*768*2);
    _Float16* w116a = (_Float16*)alloc((size_t)768*3072*2);
    _Float16* w216a = (_Float16*)alloc((size_t)3072*768*2);
    _Float16* wq16d = (_Float16*)alloc((size_t)768*768*2);
    _Float16* wkv16d= (_Float16*)alloc((size_t)768*1536*2);
    _Float16* wo16d = (_Float16*)alloc((size_t)768*768*2);
    _Float16* w116d = (_Float16*)alloc((size_t)768*3072*2);
    _Float16* w216d = (_Float16*)alloc((size_t)3072*768*2);
    _Float16* fcwp  = (_Float16*)alloc((size_t)800*768*2);
    _Float16* cat   = kvA;   // alias: kvA (201MB) dead after agg attention; cat needs 105MB

    auto f2h = [&](const float* s, _Float16* d, int n){
        k_f2h<<<dim3((n + 255)/256), dim3(256), 0, stream>>>(s, d, n);
    };
    k_zero<<<1, 32, 0, stream>>>(accs);
    f2h(awq,  wq16a,  768*768);   f2h(awkv, wkv16a, 768*1536);
    f2h(awo,  wo16a,  768*768);   f2h(aw1,  w116a,  768*3072);
    f2h(aw2,  w216a,  3072*768);
    f2h(dwq,  wq16d,  768*768);   f2h(dwkv, wkv16d, 768*1536);
    f2h(dwo,  wo16d,  768*768);   f2h(dw1,  w116d,  768*3072);
    f2h(dw2,  w216d,  3072*768);
    k_fcwpad<<<(800*768 + 255)/256, 256, 0, stream>>>(fcw, fcwp);
    k_rep_proto<<<(96*768 + 255)/256, 256, 0, stream>>>(proto, prep);
    k_down<<<4096, 256, 0, stream>>>(x, x0);

    // ---- aggregation block (6 prototypes attend to 4096 tokens) ----
    k_ln<<<65536, 256, 0, stream>>>(x, ag1, ab1, xn);
    k_gemm<EB|EH, true ><<<dim3(12, 512), 256, 0, stream>>>(xn, wkv16a, abkv, nullptr, kvA, 65536, 1536, 768);
    k_ln<<<6, 256, 0, stream>>>(proto, ag1, ab1, pln);
    k_gemm<EB|EH, false><<<dim3(6, 1), 256, 0, stream>>>(pln, wq16a, abq, nullptr, qhA, 6, 768, 768);
    k_agg_attn<<<192, 384, 0, stream>>>(qhA, kvA, oA);
    k_gemm<EB|ER, false><<<dim3(6, 1), 256, 0, stream>>>(oA, wo16a, abo, prep, hA, 96, 768, 768);
    k_ln<<<96, 256, 0, stream>>>(hA, ag2, ab2, hnA);
    k_gemm<EB|EG|EH, false><<<dim3(24, 1), 256, 0, stream>>>(hnA, w116a, ac1, nullptr, f1A, 96, 3072, 768);
    k_gemm<EB|ER, false><<<dim3(6, 1), 256, 0, stream>>>(f1A, w216a, ac2, hA, agg, 96, 768, 3072);

    // ---- gather loss ----
    k_gather<<<4096, 256, 0, stream>>>(x0, agg, accs);

    // ---- decoder block (x0 tokens attend to 6 aggregated prototypes) ----
    k_ln<<<96, 256, 0, stream>>>(agg, dg1, db1, aggn);
    k_ln<<<4096, 256, 0, stream>>>(x0, dg1, db1, x0n);
    k_gemm<EB|EH, true ><<<dim3(6, 32), 256, 0, stream>>>(x0n, wq16d, dbq, nullptr, qhD, 4096, 768, 768);
    k_gemm<EB|EH, false><<<dim3(12, 1), 256, 0, stream>>>(aggn, wkv16d, dbkv, nullptr, kvD, 96, 1536, 768);
    k_dec_attn<<<192, 256, 0, stream>>>(qhD, kvD, oD);
    k_gemm<EB|ER, true ><<<dim3(6, 32), 256, 0, stream>>>(oD, wo16d, dbo, x0, hD, 4096, 768, 768);
    k_ln<<<4096, 256, 0, stream>>>(hD, dg2, db2, hnD);
    k_gemm<EB|EG|EH, true ><<<dim3(24, 32), 256, 0, stream>>>(hnD, w116d, dc1, nullptr, f1D, 4096, 3072, 768);
    k_gemm<EB|ER, true ><<<dim3(6, 32), 256, 0, stream>>>(f1D, w216d, dc2, hD, de, 4096, 768, 3072);

    // ---- losses, cosine map, final FC ----
    k_sm<<<16, 256, 0, stream>>>(x0, de, accs + 1);
    k_cosmap<<<4096, 256, 0, stream>>>(x0, de, mtok);
    k_upcat<<<65536, 256, 0, stream>>>(x, mtok, cat);
    k_gemm<EB, true ><<<dim3(6, 512), 256, 0, stream>>>(cat, fcwp, fcb, nullptr, out, 65536, 768, 800);
    k_final<<<1, 32, 0, stream>>>(accs, out);
}
